// xLSTM_48584670053147
// MI455X (gfx1250) — matrix operand order repaired and checked by reading, not compiled, not run on hardware
//
#include <hip/hip_runtime.h>

typedef __attribute__((ext_vector_type(16))) __bf16 v16bf;
typedef __attribute__((ext_vector_type(8)))  float  v8f;

#define DEV static __device__ __forceinline__

// ---- constants from the reference ----
#define BB 4
#define SS 1024
#define DD 1024
#define II 2048      // I = 2*D
#define NBq 512      // NB = I/4
#define NHMh 4
#define DHm 512      // I/NHM
#define NHSh 4
#define DHSs 256
#define UU 1344
#define ROWS (BB*SS) // 4096

// =====================================================================
// WMMA helpers (CDNA5 bf16 16x16x32, f32 accumulate)
// =====================================================================
DEV v8f wmma_bf(v16bf a, v16bf b, v8f c) {
  return __builtin_amdgcn_wmma_f32_16x16x32_bf16(false, a, false, b, (short)0, c, false, false);
}

// A-operand: 16x32 bf16 tile, row-major in LDS with leading dim ld.
// lanes 0-15: M=lane, K 0..15 in e0..15
// lanes16-31: M=lane-16, K 16..31 in e0..15
DEV v16bf load_a_lds(const __bf16* t, int ld) {
  const int l = threadIdx.x & 31;
  const int m = l & 15;
  const int kb = (l >> 4) ? 16 : 0;
  v16bf a;
#pragma unroll
  for (int e = 0; e < 8; ++e) a[e] = t[m * ld + kb + e];
#pragma unroll
  for (int e = 0; e < 8; ++e) a[8 + e] = t[m * ld + kb + 8 + e];
  return a;
}

// B-operand: 32x16 bf16 (K x N). Stored in LDS as rows of N (i.e. Bs[n][k]).
// lanes 0-15: N=lane, K 0..15 ; lanes16-31: N=lane-16, K 16..31
DEV v16bf load_b_lds(const __bf16* t, int ld) {
  const int l = threadIdx.x & 31;
  const int n = l & 15;
  const int ko = (l >> 4) ? 16 : 0;
  v16bf b;
#pragma unroll
  for (int e = 0; e < 16; ++e) b[e] = t[n * ld + ko + e];
  return b;
}

// B-operand from global, element (k,n) = base[n*rowstride + kcol + k]
// (used for Q@K^T: each lane n holds one K-row)
DEV v16bf load_b_grows(const float* base, int rowstride, int kcol) {
  const int l = threadIdx.x & 31;
  const int n = l & 15;
  const int ko = (l >> 4) ? 16 : 0;
  const float* p = base + (size_t)n * rowstride + kcol + ko;
  v16bf b;
#pragma unroll
  for (int e = 0; e < 16; ++e) b[e] = (__bf16)p[e];
  return b;
}

// B-operand from global, element (k,n) = base[k*rowstride + n]
// (used for P@V: each lane n holds one V-column)
DEV v16bf load_b_gcols(const float* base, int rowstride) {
  const int l = threadIdx.x & 31;
  const int n = l & 15;
  const int ko = (l >> 4) ? 16 : 0;
  v16bf b;
#pragma unroll
  for (int e = 0; e < 16; ++e) b[e] = (__bf16)base[(size_t)(ko + e) * rowstride + n];
  return b;
}

DEV float siluf(float x) { return x / (1.f + expf(-x)); }
DEV float sigmf(float x) { return 1.f / (1.f + expf(-x)); }
DEV float logsigf(float x) {
  return (x >= 0.f) ? -log1pf(expf(-x)) : x - log1pf(expf(x));
}
DEV float geluf(float x) { return 0.5f * x * (1.f + erff(x * 0.70710678118654752f)); }

// =====================================================================
// Embedding gather
// =====================================================================
__global__ __launch_bounds__(256) void embed_kernel(const int* ids, const float* emb, float* x) {
  int i = blockIdx.x * 256 + threadIdx.x;   // over ROWS*DD
  int r = i >> 10, c = i & 1023;
  x[i] = emb[(size_t)ids[r] * DD + c];
}

// =====================================================================
// Generic (layer/multi-head) norm: rows x (nh*dh), weight length nh*dh
// =====================================================================
__global__ __launch_bounds__(256) void norm_kernel(const float* __restrict__ x, int ldx,
                                                   const float* __restrict__ w,
                                                   float* __restrict__ out, int ldo,
                                                   int nh, int dh) {
  const int row = blockIdx.x;
  const int tid = threadIdx.x;
  const int tph = 256 / nh;
  const int head = tid / tph;
  const int ln = tid % tph;
  const float* xr = x + (size_t)row * ldx + head * dh;
  float s = 0.f, s2 = 0.f;
  for (int c = ln; c < dh; c += tph) { float v = xr[c]; s += v; s2 += v * v; }
  __shared__ float sa[256], sb[256];
  sa[tid] = s; sb[tid] = s2;
  __syncthreads();
  for (int off = tph >> 1; off > 0; off >>= 1) {
    if (ln < off) { sa[tid] += sa[tid + off]; sb[tid] += sb[tid + off]; }
    __syncthreads();
  }
  float mu = sa[head * tph] / (float)dh;
  float var = sb[head * tph] / (float)dh - mu * mu;
  float inv = rsqrtf(var + 1e-5f);
  for (int c = ln; c < dh; c += tph)
    out[(size_t)row * ldo + head * dh + c] = (xr[c] - mu) * inv * w[head * dh + c];
}

// =====================================================================
// Generic GEMM: C[M,N] = A[M,K] @ B[N,K]^T (+bias[n]) (+R[m,n])
// 64x64 tile / WG, 8 waves, bf16 WMMA, 64-wide K stage (4 WMMA per stage
// per wave between barriers). M%64==0, N%64==0, K%64==0 for all call sites
// (K in {256,1024,1344? no -> 1344%64=0? 1344=21*64 yes, 2048}).
// =====================================================================
__global__ __launch_bounds__(256) void gemm_kernel(const float* __restrict__ A, int lda,
                                                   const float* __restrict__ Bw, int ldb,
                                                   float* __restrict__ C, int ldc,
                                                   const float* __restrict__ bias,
                                                   const float* __restrict__ R, int ldr,
                                                   int K) {
  __shared__ __bf16 As[64 * 64];
  __shared__ __bf16 Bs[64 * 64];
  const int tid = threadIdx.x, lane = tid & 31, wid = tid >> 5;
  const int m0 = blockIdx.y * 64, n0 = blockIdx.x * 64;
  const int mw = (wid & 3) * 16, nw = (wid >> 2) * 32;
  v8f acc0 = {}, acc1 = {};
  const int r = tid >> 2, c0 = (tid & 3) * 16;
  for (int kk = 0; kk < K; kk += 64) {
    const float* ap = A + (size_t)(m0 + r) * lda + kk + c0;
    const float* bp = Bw + (size_t)(n0 + r) * ldb + kk + c0;
    if (kk + 64 < K) {                       // prefetch next K-slab
      __builtin_prefetch(ap + 64, 0, 1);
      __builtin_prefetch(bp + 64, 0, 1);
    }
#pragma unroll
    for (int e = 0; e < 16; ++e) {
      As[r * 64 + c0 + e] = (__bf16)ap[e];
      Bs[r * 64 + c0 + e] = (__bf16)bp[e];
    }
    __syncthreads();
#pragma unroll
    for (int k2 = 0; k2 < 64; k2 += 32) {
      v16bf a = load_a_lds(As + mw * 64 + k2, 64);
      v16bf b0 = load_b_lds(Bs + nw * 64 + k2, 64);
      v16bf b1 = load_b_lds(Bs + (nw + 16) * 64 + k2, 64);
      acc0 = wmma_bf(a, b0, acc0);
      acc1 = wmma_bf(a, b1, acc1);
    }
    __syncthreads();
  }
  const int hi = lane >> 4, n = lane & 15;
#pragma unroll
  for (int t = 0; t < 2; ++t) {
    v8f acc = t ? acc1 : acc0;
#pragma unroll
    for (int rr = 0; rr < 8; ++rr) {
      int row = m0 + mw + rr + hi * 8;
      int col = n0 + nw + t * 16 + n;
      float v = acc[rr];
      if (bias) v += bias[col];
      if (R) v += R[(size_t)row * ldr + col];
      C[(size_t)row * ldc + col] = v;
    }
  }
}

// =====================================================================
// Causal depthwise conv (K=4) + bias + SiLU
// =====================================================================
__global__ __launch_bounds__(256) void conv_silu_kernel(const float* __restrict__ x, int ldx, int C,
                                                        const float* __restrict__ w,
                                                        const float* __restrict__ bias,
                                                        float* __restrict__ out) {
  int i = blockIdx.x * 256 + threadIdx.x;     // over ROWS*C
  int c = i % C;
  int rs = i / C;
  int s = rs & (SS - 1);
  int base = rs - s;                          // b*SS
  float acc = bias[c];
#pragma unroll
  for (int j = 0; j < 4; ++j) {
    int sp = s - 3 + j;
    if (sp >= 0) acc += x[(size_t)(base + sp) * ldx + c] * w[c * 4 + j];
  }
  out[(size_t)rs * C + c] = siluf(acc);
}

// =====================================================================
// Block-diagonal 4x4 headwise projection (mLSTM q/k/v)
// =====================================================================
__global__ __launch_bounds__(256) void headwise_kernel(const float* __restrict__ x, int ldx,
                                                       const float* __restrict__ w,
                                                       float* __restrict__ out, int ldo, int ooff) {
  int i = blockIdx.x * 256 + threadIdx.x;     // over ROWS*II
  int row = i >> 11, j = i & 2047;
  int nb = j >> 2, o = j & 3;
  const float* xr = x + (size_t)row * ldx + nb * 4;
  const float* wr = w + (size_t)(nb * 4 + o) * 4;
  float acc = 0.f;
#pragma unroll
  for (int k = 0; k < 4; ++k) acc += xr[k] * wr[k];
  out[(size_t)row * ldo + ooff + j] = acc;
}

// =====================================================================
// ig/fg gate projections: 8 dot products of length 3I per token
// =====================================================================
__global__ __launch_bounds__(256) void gates_kernel(const float* __restrict__ qkv,
                                                    const float* __restrict__ igw,
                                                    const float* __restrict__ igb,
                                                    const float* __restrict__ fgw,
                                                    const float* __restrict__ fgb,
                                                    float* __restrict__ ig, float* __restrict__ fg) {
  const int row = blockIdx.x;                 // b*SS + s
  const int tid = threadIdx.x, lane = tid & 31, wid = tid >> 5;
  float p[8] = {0.f, 0.f, 0.f, 0.f, 0.f, 0.f, 0.f, 0.f};
  const float* qr = qkv + (size_t)row * (3 * II);
  for (int c = tid; c < 3 * II; c += 256) {
    float v = qr[c];
#pragma unroll
    for (int h = 0; h < 4; ++h) {
      p[h]     += v * igw[h * (3 * II) + c];
      p[4 + h] += v * fgw[h * (3 * II) + c];
    }
  }
  __shared__ float red[8 * 8];
#pragma unroll
  for (int g = 0; g < 8; ++g)
    for (int off = 16; off > 0; off >>= 1) p[g] += __shfl_down(p[g], off);
  if (lane == 0)
#pragma unroll
    for (int g = 0; g < 8; ++g) red[wid * 8 + g] = p[g];
  __syncthreads();
  if (tid < 8) {
    float s = 0.f;
#pragma unroll
    for (int w2 = 0; w2 < 8; ++w2) s += red[w2 * 8 + tid];
    int h = tid & 3;
    int b = row >> 10, sidx = row & (SS - 1);
    if (tid < 4) ig[(size_t)(b * 4 + h) * SS + sidx] = s + igb[h];
    else         fg[(size_t)(b * 4 + h) * SS + sidx] = s + fgb[h];
  }
}

// =====================================================================
// Per-(b,h) serial gate scan: csum, prefix-max -> rowconst/colconst/nlow
// =====================================================================
__global__ void scan_gates_kernel(const float* __restrict__ ig, const float* __restrict__ fg,
                                  float* __restrict__ rowc, float* __restrict__ colc,
                                  float* __restrict__ nlow) {
  if (threadIdx.x != 0) return;
  int bh = blockIdx.x;
  const float* igp = ig + (size_t)bh * SS;
  const float* fgp = fg + (size_t)bh * SS;
  float* rc = rowc + (size_t)bh * SS;
  float* cc = colc + (size_t)bh * SS;
  float* nl = nlow + (size_t)bh * SS;
  float csum = 0.f, runmax = -3.4e38f;
  for (int s = 0; s < SS; ++s) {
    csum += logsigf(fgp[s]);
    float e = igp[s] - csum;
    cc[s] = e;
    runmax = fmaxf(runmax, e);
    float maxD = csum + runmax;
    rc[s] = csum - maxD;
    nl[s] = expf(-maxD);
  }
}

// =====================================================================
// mLSTM fused attention (flash-style, per (b,h,16-row s-tile))
// qkv layout per row: [q(2048) | k(2048) | v(2048)], head h = 512-wide slice.
// =====================================================================
__global__ __launch_bounds__(256) void mlstm_attn_kernel(const float* __restrict__ qkv,
                                                         const float* __restrict__ rowc,
                                                         const float* __restrict__ colc,
                                                         const float* __restrict__ nlow,
                                                         float* __restrict__ out) {
  const int st = blockIdx.x, h = blockIdx.y, b = blockIdx.z;
  const int tid = threadIdx.x, lane = tid & 31, wid = tid >> 5;
  const int s0 = st * 16, s_hi = s0 + 15;
  const int LQ = 3 * II;   // 6144

  __shared__ __bf16 qs[16 * 512];
  __shared__ __bf16 ps[16 * 256];
  __shared__ float scale[16];

  const float qsc = 0.04419417382415922f;   // 1/sqrt(512)
  for (int i = tid; i < 16 * 512; i += 256) {
    int r = i >> 9, c = i & 511;
    qs[i] = (__bf16)(qkv[(size_t)(b * SS + s0 + r) * LQ + h * 512 + c] * qsc);
  }
  const float* rc = rowc + (size_t)(b * 4 + h) * SS;
  const float* cc = colc + (size_t)(b * 4 + h) * SS;
  const float* nl = nlow + (size_t)(b * 4 + h) * SS;

  v8f oacc[4]; oacc[0] = v8f{}; oacc[1] = v8f{}; oacc[2] = v8f{}; oacc[3] = v8f{};
  v8f racc = {};
  v16bf bones;
#pragma unroll
  for (int e = 0; e < 16; ++e) bones[e] = (__bf16)1.0f;

  __syncthreads();

  for (int t0 = 0; t0 <= s_hi; t0 += 256) {
    const int tw = t0 + wid * 32;
    v8f sacc0 = {}, sacc1 = {};
    if (tw <= s_hi) {                                   // wave-uniform
      const float* kbase = qkv + (size_t)(b * SS + tw) * LQ + II + h * 512;
      for (int kk = 0; kk < 512; kk += 32) {
        v16bf a = load_a_lds(qs + kk, 512);
        v16bf b0 = load_b_grows(kbase, LQ, kk);
        v16bf b1 = load_b_grows(kbase + 16 * LQ, LQ, kk);
        sacc0 = wmma_bf(a, b0, sacc0);
        sacc1 = wmma_bf(a, b1, sacc1);
      }
    }
    // apply decay + causal mask, write P (bf16) to LDS
    {
      const int hi = lane >> 4, n = lane & 15;
#pragma unroll
      for (int t = 0; t < 2; ++t) {
        v8f sv = t ? sacc1 : sacc0;
#pragma unroll
        for (int r = 0; r < 8; ++r) {
          int row = r + hi * 8;
          int sg = s0 + row;
          int col = tw + t * 16 + n;
          float p = 0.f;
          if (tw <= s_hi && col <= sg) p = sv[r] * expf(rc[sg] + cc[col]);
          ps[row * 256 + wid * 32 + t * 16 + n] = (__bf16)p;
        }
      }
    }
    __syncthreads();
    // P(16x256) @ V(256x64-chunk) ; wave0 also accumulates row-sums via ones-B
    for (int ks = 0; ks < 8; ++ks) {
      int tk = t0 + ks * 32;
      v16bf a = load_a_lds(ps + ks * 32, 256);
      if (wid == 0) racc = wmma_bf(a, bones, racc);
      const float* vbase = qkv + (size_t)(b * SS + tk) * LQ + 2 * II + h * 512 + wid * 64;
#pragma unroll
      for (int nt = 0; nt < 4; ++nt) {
        v16bf bb = load_b_gcols(vbase + nt * 16, LQ);
        oacc[nt] = wmma_bf(a, bb, oacc[nt]);
      }
    }
    __syncthreads();
  }
  if (wid == 0 && (lane & 15) == 0) {
    const int hi = lane >> 4;
#pragma unroll
    for (int r = 0; r < 8; ++r) {
      int row = r + hi * 8;
      float nrm = fmaxf(fabsf(racc[r]), nl[s0 + row]) + 1e-6f;
      scale[row] = 1.0f / nrm;
    }
  }
  __syncthreads();
  {
    const int hi = lane >> 4, n = lane & 15;
#pragma unroll
    for (int nt = 0; nt < 4; ++nt)
#pragma unroll
      for (int r = 0; r < 8; ++r) {
        int row = r + hi * 8;
        out[(size_t)(b * SS + s0 + row) * II + h * 512 + wid * 64 + nt * 16 + n] =
            oacc[nt][r] * scale[row];
      }
  }
}

// =====================================================================
// mLSTM combine: y = (hn + skip*xa) * silu(z)
// =====================================================================
__global__ __launch_bounds__(256) void mlstm_combine_kernel(const float* __restrict__ hn,
                                                            const float* __restrict__ xa,
                                                            const float* __restrict__ up,
                                                            const float* __restrict__ skip,
                                                            float* __restrict__ y) {
  int i = blockIdx.x * 256 + threadIdx.x;     // over ROWS*II
  int row = i >> 11, c = i & 2047;
  float z = up[(size_t)row * 4096 + II + c];
  y[i] = (hn[i] + skip[c] * xa[i]) * siluf(z);
}

// =====================================================================
// sLSTM sequential scan: one WG (1024 thr) per head, all 4 batches inside
// =====================================================================
__global__ __launch_bounds__(1024) void slstm_scan_kernel(const float* __restrict__ prei,
                                                          const float* __restrict__ pref,
                                                          const float* __restrict__ prez,
                                                          const float* __restrict__ preo,
                                                          const float* __restrict__ rk,
                                                          const float* __restrict__ rb,
                                                          float* __restrict__ ys) {
  const int h = blockIdx.x;
  const int tid = threadIdx.x;
  const int g = tid >> 8, d = tid & 255;      // phase-1 role
  const int b2 = tid >> 8, d2 = tid & 255;    // phase-2 role
  __shared__ float hprev[4][256];
  __shared__ float raw[16][256];              // [b*4+g][d]
  hprev[b2][d2] = 0.f;
  float c = 0.f, nn = 0.f, m = 0.f;
  const float* pre[4] = {prei, pref, prez, preo};
  const float* rkb = rk + (size_t)h * 262144 + g * 256 + d;   // stride 1024 over k
  const float rbv = rb[(size_t)(h * 4 + g) * 256 + d];
  __syncthreads();
  for (int s = 0; s < SS; ++s) {
    // phase 1: raw[b][g][d] = gt + h_prev @ rk + rb
    float acc0 = pre[g][(size_t)(0 * SS + s) * DD + h * 256 + d] + rbv;
    float acc1 = pre[g][(size_t)(1 * SS + s) * DD + h * 256 + d] + rbv;
    float acc2 = pre[g][(size_t)(2 * SS + s) * DD + h * 256 + d] + rbv;
    float acc3 = pre[g][(size_t)(3 * SS + s) * DD + h * 256 + d] + rbv;
    for (int k = 0; k < 256; ++k) {
      float w = rkb[(size_t)k * 1024];
      acc0 += hprev[0][k] * w;
      acc1 += hprev[1][k] * w;
      acc2 += hprev[2][k] * w;
      acc3 += hprev[3][k] * w;
    }
    raw[0 * 4 + g][d] = acc0;
    raw[1 * 4 + g][d] = acc1;
    raw[2 * 4 + g][d] = acc2;
    raw[3 * 4 + g][d] = acc3;
    __syncthreads();
    // phase 2: gate math, state update (per (b,d) thread)
    float ir = raw[b2 * 4 + 0][d2];
    float fr = raw[b2 * 4 + 1][d2];
    float zr = raw[b2 * 4 + 2][d2];
    float orr = raw[b2 * 4 + 3][d2];
    float lf = m + logsigf(fr);
    float mn = fmaxf(ir, lf);
    float ih = expf(ir - mn);
    float fh = expf(lf - mn);
    c = fh * c + ih * tanhf(zr);
    nn = fh * nn + ih;
    m = mn;
    float hv = sigmf(orr) * c / nn;
    hprev[b2][d2] = hv;
    ys[(size_t)(b2 * SS + s) * DD + h * 256 + d2] = hv;
    __syncthreads();
  }
}

// =====================================================================
// FFN combine: t = gelu(gate) * up   (gate = up[:,:U], upp = up[:,U:])
// =====================================================================
__global__ __launch_bounds__(256) void ffn_combine_kernel(const float* __restrict__ up,
                                                          float* __restrict__ t) {
  int i = blockIdx.x * 256 + threadIdx.x;     // over ROWS*UU
  int row = i / UU, cc = i % UU;
  float gate = up[(size_t)row * (2 * UU) + cc];
  float upp = up[(size_t)row * (2 * UU) + UU + cc];
  t[i] = geluf(gate) * upp;
}

__global__ __launch_bounds__(256) void add_kernel(float* __restrict__ x,
                                                  const float* __restrict__ y) {
  int i = blockIdx.x * 256 + threadIdx.x;
  x[i] += y[i];
}

// =====================================================================
// Host orchestration
// =====================================================================
extern "C" void kernel_launch(void* const* d_in, const int* in_sizes, int n_in,
                              void* d_out, int out_size, void* d_ws, size_t ws_size,
                              hipStream_t stream) {
  (void)in_sizes; (void)n_in; (void)out_size; (void)ws_size;
  const int*   ids        = (const int*)d_in[0];
  const float* emb        = (const float*)d_in[1];
  const float* m_ln_w     = (const float*)d_in[2];
  const float* m_up_w     = (const float*)d_in[3];
  const float* m_conv_w   = (const float*)d_in[4];
  const float* m_conv_b   = (const float*)d_in[5];
  const float* m_q_w      = (const float*)d_in[6];
  const float* m_k_w      = (const float*)d_in[7];
  const float* m_v_w      = (const float*)d_in[8];
  const float* m_ig_w     = (const float*)d_in[9];
  const float* m_ig_b     = (const float*)d_in[10];
  const float* m_fg_w     = (const float*)d_in[11];
  const float* m_fg_b     = (const float*)d_in[12];
  const float* m_skip     = (const float*)d_in[13];
  const float* m_on_w     = (const float*)d_in[14];
  const float* m_down_w   = (const float*)d_in[15];
  const float* s_ln_w     = (const float*)d_in[16];
  const float* s_conv_w   = (const float*)d_in[17];
  const float* s_conv_b   = (const float*)d_in[18];
  const float* s_iw       = (const float*)d_in[19];
  const float* s_fw       = (const float*)d_in[20];
  const float* s_zw       = (const float*)d_in[21];
  const float* s_ow       = (const float*)d_in[22];
  const float* s_rk       = (const float*)d_in[23];
  const float* s_rb       = (const float*)d_in[24];
  const float* s_gn_w     = (const float*)d_in[25];
  const float* s_ln2_w    = (const float*)d_in[26];
  const float* s_ffn_up   = (const float*)d_in[27];
  const float* s_ffn_down = (const float*)d_in[28];
  const float* post_ln_w  = (const float*)d_in[29];
  const float* head_w     = (const float*)d_in[30];
  const float* head_b     = (const float*)d_in[31];

  float* ws  = (float*)d_ws;
  float* X   = ws;                       // 4096x1024
  float* XN  = ws + 4194304;             // 4096x1024
  float* UP  = ws + 8388608;             // 4096x4096
  float* XA  = ws + 25165824;            // 4096x2048
  float* QKV = ws + 33554432;            // 4096x6144 (also Y / sLSTM pre-gates)
  float* H   = ws + 58720256;            // 4096x2048 (also ys / ffn product)
  float* HN  = ws + 67108864;            // 4096x2048
  float* IG  = ws + 75497472;            // 16K
  float* FG  = IG + 16384;
  float* RC  = FG + 16384;
  float* CC  = RC + 16384;
  float* NL  = CC + 16384;

  auto gemm = [&](const float* A, int lda, const float* B, int ldb, float* C, int ldc,
                  const float* bias, const float* R, int ldr, int M, int N, int K) {
    gemm_kernel<<<dim3(N / 64, M / 64), 256, 0, stream>>>(A, lda, B, ldb, C, ldc, bias, R, ldr, K);
  };

  embed_kernel<<<(ROWS * DD) / 256, 256, 0, stream>>>(ids, emb, X);

  int mi = 0;
  for (int blk = 0; blk < 4; ++blk) {
    if (blk == 1) {
      // ---------------- sLSTM block ----------------
      norm_kernel<<<ROWS, 256, 0, stream>>>(X, DD, s_ln_w, XN, DD, 1, DD);
      conv_silu_kernel<<<(ROWS * DD) / 256, 256, 0, stream>>>(XN, DD, DD, s_conv_w, s_conv_b, XA);
      for (int g = 0; g < 4; ++g) {
        const float* W = (g == 0) ? s_iw : (g == 1) ? s_fw : (g == 2) ? s_zw : s_ow;
        const float* IN = (g < 2) ? XA : XN;
        float* PRE = QKV + (size_t)g * 4194304;
        for (int hh = 0; hh < 4; ++hh)
          gemm(IN + hh * 256, DD, W + (size_t)hh * 65536, 256, PRE + hh * 256, DD,
               nullptr, nullptr, 0, ROWS, 256, 256);
      }
      slstm_scan_kernel<<<4, 1024, 0, stream>>>(QKV, QKV + 4194304, QKV + 2 * 4194304,
                                                QKV + 3 * 4194304, s_rk, s_rb, H);
      norm_kernel<<<ROWS, 256, 0, stream>>>(H, DD, s_gn_w, HN, DD, 4, 256);
      add_kernel<<<(ROWS * DD) / 256, 256, 0, stream>>>(X, HN);
      norm_kernel<<<ROWS, 256, 0, stream>>>(X, DD, s_ln2_w, XN, DD, 1, DD);
      gemm(XN, DD, s_ffn_up, DD, UP, 2 * UU, nullptr, nullptr, 0, ROWS, 2 * UU, DD);
      ffn_combine_kernel<<<(ROWS * UU) / 256, 256, 0, stream>>>(UP, H);
      gemm(H, UU, s_ffn_down, UU, X, DD, nullptr, X, DD, ROWS, DD, UU);
    } else {
      // ---------------- mLSTM block ----------------
      const float* lnw   = m_ln_w + (size_t)mi * DD;
      const float* upw   = m_up_w + (size_t)mi * 4096 * DD;
      const float* convw = m_conv_w + (size_t)mi * II * 4;
      const float* convb = m_conv_b + (size_t)mi * II;
      const float* qw    = m_q_w + (size_t)mi * NBq * 16;
      const float* kw    = m_k_w + (size_t)mi * NBq * 16;
      const float* vw    = m_v_w + (size_t)mi * NBq * 16;
      const float* igw   = m_ig_w + (size_t)mi * 4 * (3 * II);
      const float* igb   = m_ig_b + (size_t)mi * 4;
      const float* fgw   = m_fg_w + (size_t)mi * 4 * (3 * II);
      const float* fgb   = m_fg_b + (size_t)mi * 4;
      const float* skp   = m_skip + (size_t)mi * II;
      const float* onw   = m_on_w + (size_t)mi * II;
      const float* dw    = m_down_w + (size_t)mi * DD * II;

      norm_kernel<<<ROWS, 256, 0, stream>>>(X, DD, lnw, XN, DD, 1, DD);
      gemm(XN, DD, upw, DD, UP, 4096, nullptr, nullptr, 0, ROWS, 4096, DD);
      conv_silu_kernel<<<(ROWS * II) / 256, 256, 0, stream>>>(UP, 4096, II, convw, convb, XA);
      headwise_kernel<<<(ROWS * II) / 256, 256, 0, stream>>>(XA, II, qw, QKV, 3 * II, 0);
      headwise_kernel<<<(ROWS * II) / 256, 256, 0, stream>>>(XA, II, kw, QKV, 3 * II, II);
      headwise_kernel<<<(ROWS * II) / 256, 256, 0, stream>>>(UP, 4096, vw, QKV, 3 * II, 2 * II);
      gates_kernel<<<ROWS, 256, 0, stream>>>(QKV, igw, igb, fgw, fgb, IG, FG);
      scan_gates_kernel<<<16, 32, 0, stream>>>(IG, FG, RC, CC, NL);
      mlstm_attn_kernel<<<dim3(SS / 16, NHMh, BB), 256, 0, stream>>>(QKV, RC, CC, NL, H);
      norm_kernel<<<ROWS, 256, 0, stream>>>(H, II, onw, HN, II, 4, 512);
      mlstm_combine_kernel<<<(ROWS * II) / 256, 256, 0, stream>>>(HN, XA, UP, skp, QKV);
      gemm(QKV, II, dw, II, X, DD, nullptr, X, DD, ROWS, DD, II);
      ++mi;
    }
  }
  norm_kernel<<<ROWS, 256, 0, stream>>>(X, DD, post_ln_w, XN, DD, 1, DD);
  gemm(XN, DD, head_w, DD, (float*)d_out, 256, head_b, nullptr, 0, ROWS, 256, DD);
}
